// VariationalGCNEncoder_S2_54065048322430
// MI455X (gfx1250) — matrix operand order repaired and checked by reading, not compiled, not run on hardware
//
#include <hip/hip_runtime.h>
#include <hip/hip_bf16.h>

typedef __attribute__((ext_vector_type(16))) _Float16 v16h;
typedef __attribute__((ext_vector_type(8)))  float    v8f;

// ---------------- problem constants ----------------
constexpr int    NN    = 262144;
constexpr int    EIN   = 4194304;
constexpr int    ETOT  = EIN + NN;          // edges + self loops
constexpr int    IN_F  = 32;
constexpr int    HEADS = 3, E1F = 15, E2F = 9, L1F = 100, L2F = 4, OUTF = 128, NMOL = 256;
constexpr int    HF    = HEADS * E1F;       // 45
constexpr int    BROWS = NN / NMOL;         // 1024
constexpr int    BCOLS = NMOL * L2F;        // 1024

// ---------------- workspace layout (floats) ----------------
constexpr size_t OFF_H   = 0;                                 // [NN,45]
constexpr size_t OFF_AS  = OFF_H   + (size_t)NN * HF;         // [NN,3]
constexpr size_t OFF_AD  = OFF_AS  + (size_t)NN * HEADS;      // [NN,3]
constexpr size_t OFF_MK  = OFF_AD  + (size_t)NN * HEADS;      // [NN,3] uint max-keys
constexpr size_t OFF_SS  = OFF_MK  + (size_t)NN * HEADS;      // [NN,3] softmax sums
constexpr size_t OFF_DEG = OFF_SS  + (size_t)NN * HEADS;      // [NN]
constexpr size_t OFF_AGG = OFF_DEG + (size_t)NN;              // [NN,45]
constexpr size_t OFF_XG  = OFF_AGG + (size_t)NN * HF;         // [NN,15]
constexpr size_t OFF_HC  = OFF_XG  + (size_t)NN * E1F;        // [NN,9]
constexpr size_t OFF_XC  = OFF_HC  + (size_t)NN * E2F;        // [NN,9]
constexpr size_t OFF_DIN = OFF_XC  + (size_t)NN * E2F;        // [NN]
constexpr size_t OFF_Z   = OFF_DIN + (size_t)NN;              // [NN,4] == [1024,1024]
constexpr size_t OFF_Z2  = OFF_Z   + (size_t)NN * L2F;        // [1024,128]
constexpr size_t OFF_BN  = OFF_Z2  + (size_t)BROWS * OUTF;    // mean1[15] is1[15] mean2[9] is2[9]

// ---------------- helpers ----------------
__device__ __forceinline__ float lrelu(float x, float a) { return x >= 0.f ? x : a * x; }

// monotone float->uint key so atomicMax(uint) == float max; 0 < key(-inf)
__device__ __forceinline__ unsigned fkey(float f) {
  unsigned u = __float_as_uint(f);
  return (u & 0x80000000u) ? ~u : (u | 0x80000000u);
}
__device__ __forceinline__ float fkey_inv(unsigned k) {
  unsigned u = (k & 0x80000000u) ? (k & 0x7FFFFFFFu) : ~k;
  return __uint_as_float(u);
}

__device__ __forceinline__ void edge_pair(const int* __restrict__ ei, int i, int& s, int& d) {
  if (i < EIN) { s = ei[i]; d = ei[EIN + i]; }
  else         { s = i - EIN; d = s; }           // appended self loops
}

// A fragment: 16x32 tile of f32 row-major [*, ld], converted to f16.
// k order used here (A and B agree): lanes 0-15 row M=lane, K=0..15; lanes 16-31 K=16..31.
__device__ __forceinline__ v16h load_a_f32(const float* __restrict__ A, int row0, int ld,
                                           int k0, int lane) {
  const int row = row0 + (lane & 15);
  const int kb  = (lane < 16) ? 0 : 16;
  const float* base = A + (size_t)row * ld + k0;
  v16h a;
#pragma unroll
  for (int v = 0; v < 8; ++v) {
    int k = kb + 2 * v;   // pairs {k,k+1}
    a[2 * v]     = (_Float16)base[k];
    a[2 * v + 1] = (_Float16)base[k + 1];
  }
  return a;
}

// B fragment: 32x16 tile of f32 row-major [K, ld] at (k0, n0).
// k order used here (same as A): lanes 0-15 hold K=0..15, lanes 16-31 K=16..31; VGPR v = pair {2v,2v+1}.
// BOUNDED: branch-free column padding — clamp index (always-valid address),
// unconditional load, then v_cndmask the value to zero. No exec-mask ping-pong.
template <bool BOUNDED>
__device__ __forceinline__ v16h load_b_f32(const float* __restrict__ B, int k0, int ld,
                                           int n0, int ncols, int lane) {
  const int n  = n0 + (lane & 15);
  const bool ok = !BOUNDED || (n < ncols);
  const int nc  = BOUNDED ? (ok ? n : (ncols - 1)) : n;    // clamped, always in-bounds
  const int kb  = (lane < 16) ? 0 : 16;
  v16h b;
#pragma unroll
  for (int v = 0; v < 8; ++v) {
    int k = k0 + kb + 2 * v;
    float x0 = B[(size_t)k * ld + nc];
    float x1 = B[(size_t)(k + 1) * ld + nc];
    if (BOUNDED) { x0 = ok ? x0 : 0.f; x1 = ok ? x1 : 0.f; }
    b[2 * v]     = (_Float16)x0;
    b[2 * v + 1] = (_Float16)x1;
  }
  return b;
}

// ---------------- kernels ----------------
__global__ void k_zero(float* __restrict__ p, size_t n) {
  for (size_t i = (size_t)blockIdx.x * blockDim.x + threadIdx.x; i < n;
       i += (size_t)gridDim.x * blockDim.x)
    p[i] = 0.f;
}

// h = x @ w_gat   (262144x32 @ 32x45), one wave per 16-row tile, 3 WMMAs (N padded to 48)
__global__ void k_gat_proj(const float* __restrict__ x, const float* __restrict__ wgat,
                           float* __restrict__ h) {
  const int lane = threadIdx.x & 31;
  const int wave = (blockIdx.x * blockDim.x + threadIdx.x) >> 5;
  const int row0 = wave * 16;
  if (row0 >= NN) return;                       // wave-uniform (grid exact anyway)
  v16h a = load_a_f32(x, row0, IN_F, 0, lane);
  const int col = lane & 15;
  const int mb  = (lane < 16) ? 0 : 8;
#pragma unroll
  for (int ct = 0; ct < 3; ++ct) {
    v16h b = load_b_f32<true>(wgat, 0, HF, ct * 16, HF, lane);
    v8f  c = {};
    c = __builtin_amdgcn_wmma_f32_16x16x32_f16(false, a, false, b, (short)0, c, false, false);
    int n = ct * 16 + col;
    if (n < HF) {
#pragma unroll
      for (int j = 0; j < 8; ++j)
        h[(size_t)(row0 + mb + j) * HF + n] = c[j];
    }
  }
}

// per (node, head): attention logit halves
__global__ void k_attn(const float* __restrict__ h, const float* __restrict__ asrc,
                       const float* __restrict__ adst, float* __restrict__ a_s,
                       float* __restrict__ a_d) {
  int i = blockIdx.x * blockDim.x + threadIdx.x;
  if (i >= NN * HEADS) return;
  int node = i / HEADS, hd = i % HEADS;
  const float* hr = h + (size_t)node * HF + hd * E1F;
  float s = 0.f, d = 0.f;
#pragma unroll
  for (int e = 0; e < E1F; ++e) {
    s += hr[e] * asrc[hd * E1F + e];
    d += hr[e] * adst[hd * E1F + e];
  }
  a_s[i] = s; a_d[i] = d;
}

__global__ void k_edge_max(const int* __restrict__ ei, const float* __restrict__ a_s,
                           const float* __restrict__ a_d, unsigned* __restrict__ mkey,
                           float* __restrict__ deg) {
  int i = blockIdx.x * blockDim.x + threadIdx.x;
  if (i >= ETOT) return;
  __builtin_prefetch(ei + i + 8192, 0, 0);      // stream edge_index ahead (global_prefetch_b8)
  int s, d; edge_pair(ei, i, s, d);
#pragma unroll
  for (int hd = 0; hd < HEADS; ++hd) {
    float e = lrelu(a_s[s * HEADS + hd] + a_d[d * HEADS + hd], 0.2f);
    atomicMax(&mkey[d * HEADS + hd], fkey(e));
  }
  atomicAdd(&deg[d], 1.0f);
}

__global__ void k_edge_sum(const int* __restrict__ ei, const float* __restrict__ a_s,
                           const float* __restrict__ a_d, const unsigned* __restrict__ mkey,
                           float* __restrict__ ssum) {
  int i = blockIdx.x * blockDim.x + threadIdx.x;
  if (i >= ETOT) return;
  int s, d; edge_pair(ei, i, s, d);
#pragma unroll
  for (int hd = 0; hd < HEADS; ++hd) {
    float e = lrelu(a_s[s * HEADS + hd] + a_d[d * HEADS + hd], 0.2f);
    float m = fkey_inv(mkey[d * HEADS + hd]);
    atomicAdd(&ssum[d * HEADS + hd], expf(e - m));
  }
}

__global__ void k_edge_agg(const int* __restrict__ ei, const float* __restrict__ a_s,
                           const float* __restrict__ a_d, const unsigned* __restrict__ mkey,
                           const float* __restrict__ ssum, const float* __restrict__ h,
                           float* __restrict__ agg) {
  int i = blockIdx.x * blockDim.x + threadIdx.x;
  if (i >= ETOT) return;
  int s, d; edge_pair(ei, i, s, d);
  const float* hr = h + (size_t)s * HF;
#pragma unroll
  for (int hd = 0; hd < HEADS; ++hd) {
    float e = lrelu(a_s[s * HEADS + hd] + a_d[d * HEADS + hd], 0.2f);
    float m = fkey_inv(mkey[d * HEADS + hd]);
    float alpha = expf(e - m) / (ssum[d * HEADS + hd] + 1e-16f);
#pragma unroll
    for (int e1 = 0; e1 < E1F; ++e1)
      atomicAdd(&agg[(size_t)d * HF + hd * E1F + e1], hr[hd * E1F + e1] * alpha);
  }
}

// xg = (agg + b_gat) @ w_ht + b_ht   (45 -> 15)
__global__ void k_node_ht(const float* __restrict__ agg, const float* __restrict__ bgat,
                          const float* __restrict__ wht, const float* __restrict__ bht,
                          float* __restrict__ xg) {
  int n = blockIdx.x * blockDim.x + threadIdx.x;
  if (n >= NN) return;
  float a45[HF];
#pragma unroll
  for (int k = 0; k < HF; ++k) a45[k] = agg[(size_t)n * HF + k] + bgat[k];
#pragma unroll
  for (int c = 0; c < E1F; ++c) {
    float s = bht[c];
#pragma unroll
    for (int k = 0; k < HF; ++k) s += a45[k] * wht[k * E1F + c];
    xg[(size_t)n * E1F + c] = s;
  }
}

// one block per feature: training-mode batch stats (biased var), eps=1e-5
__global__ void k_bn_reduce(const float* __restrict__ xp, int C,
                            float* __restrict__ mean, float* __restrict__ invstd) {
  __shared__ float s1[256], s2[256];
  const int c = blockIdx.x, tid = threadIdx.x;
  float a = 0.f, b = 0.f;
  for (int i = tid; i < NN; i += blockDim.x) {
    float v = xp[(size_t)i * C + c];
    a += v; b += v * v;
  }
  s1[tid] = a; s2[tid] = b; __syncthreads();
  for (int st = 128; st > 0; st >>= 1) {
    if (tid < st) { s1[tid] += s1[tid + st]; s2[tid] += s2[tid + st]; }
    __syncthreads();
  }
  if (tid == 0) {
    float m = s1[0] / (float)NN;
    float var = s2[0] / (float)NN - m * m;
    mean[c] = m; invstd[c] = rsqrtf(var + 1e-5f);
  }
}

__global__ void k_dinv(const float* __restrict__ deg, float* __restrict__ dinv) {
  int n = blockIdx.x * blockDim.x + threadIdx.x;
  if (n >= NN) return;
  float d = deg[n];
  dinv[n] = d > 0.f ? rsqrtf(d) : 0.f;
}

// hc = BN1(xg) @ w_gcn   (15 -> 9)
__global__ void k_node_gcn(const float* __restrict__ xg, const float* __restrict__ mean1,
                           const float* __restrict__ is1, const float* __restrict__ g1,
                           const float* __restrict__ be1, const float* __restrict__ wgcn,
                           float* __restrict__ hc) {
  int n = blockIdx.x * blockDim.x + threadIdx.x;
  if (n >= NN) return;
  float xb[E1F];
#pragma unroll
  for (int c = 0; c < E1F; ++c)
    xb[c] = (xg[(size_t)n * E1F + c] - mean1[c]) * is1[c] * g1[c] + be1[c];
#pragma unroll
  for (int j = 0; j < E2F; ++j) {
    float s = 0.f;
#pragma unroll
    for (int c = 0; c < E1F; ++c) s += xb[c] * wgcn[c * E2F + j];
    hc[(size_t)n * E2F + j] = s;
  }
}

__global__ void k_edge_gcn(const int* __restrict__ ei, const float* __restrict__ dinv,
                           const float* __restrict__ hc, float* __restrict__ xc) {
  int i = blockIdx.x * blockDim.x + threadIdx.x;
  if (i >= ETOT) return;
  int s, d; edge_pair(ei, i, s, d);
  float nrm = dinv[s] * dinv[d];
  const float* hr = hc + (size_t)s * E2F;
#pragma unroll
  for (int j = 0; j < E2F; ++j)
    atomicAdd(&xc[(size_t)d * E2F + j], hr[j] * nrm);
}

// BN2(xc) -> 100 -> 4 fused MLP (b_gcn cancels inside BN's mean subtraction)
__global__ void k_node_mlp(const float* __restrict__ xc, const float* __restrict__ mean2,
                           const float* __restrict__ is2, const float* __restrict__ g2,
                           const float* __restrict__ be2, const float* __restrict__ wl1,
                           const float* __restrict__ bl1, const float* __restrict__ wl2,
                           const float* __restrict__ bl2, float* __restrict__ z) {
  int n = blockIdx.x * blockDim.x + threadIdx.x;
  if (n >= NN) return;
  float xb[E2F];
#pragma unroll
  for (int c = 0; c < E2F; ++c)
    xb[c] = (xc[(size_t)n * E2F + c] - mean2[c]) * is2[c] * g2[c] + be2[c];
  float z1[L1F];
  for (int j = 0; j < L1F; ++j) {
    float s = bl1[j];
#pragma unroll
    for (int c = 0; c < E2F; ++c) s += xb[c] * wl1[c * L1F + j];
    z1[j] = lrelu(s, 0.01f);
  }
#pragma unroll
  for (int k = 0; k < L2F; ++k) {
    float s = bl2[k];
    for (int j = 0; j < L1F; ++j) s += z1[j] * wl2[j * L2F + k];
    z[(size_t)n * L2F + k] = lrelu(s, 0.01f);
  }
}

// z2 = lrelu( Z[1024,1024] @ w_tr[1024,128] + b_tr ); one wave per 16x16 output tile
__global__ void k_trans_gemm(const float* __restrict__ z, const float* __restrict__ wtr,
                             const float* __restrict__ btr, float* __restrict__ z2) {
  const int lane = threadIdx.x & 31;
  const int wave = (blockIdx.x * blockDim.x + threadIdx.x) >> 5;
  if (wave >= (BROWS / 16) * (OUTF / 16)) return;   // wave-uniform
  const int mt = wave >> 3, nt = wave & 7;
  const int row0 = mt * 16, n0 = nt * 16;
  v8f c = {};
  for (int k0 = 0; k0 < BCOLS; k0 += 32) {
    v16h a = load_a_f32(z, row0, BCOLS, k0, lane);
    v16h b = load_b_f32<false>(wtr, k0, OUTF, n0, OUTF, lane);   // unbounded: no predication
    c = __builtin_amdgcn_wmma_f32_16x16x32_f16(false, a, false, b, (short)0, c, false, false);
  }
  const int col = n0 + (lane & 15);
  const int mb  = (lane < 16) ? 0 : 8;
  const float bias = btr[col];
#pragma unroll
  for (int j = 0; j < 8; ++j) {
    float v = c[j] + bias;
    z2[(size_t)(row0 + mb + j) * OUTF + col] = lrelu(v, 0.01f);
  }
}

// mu = z2 @ w_mu + b_mu ; ls = z2 @ w_ls + b_ls
__global__ void k_heads(const float* __restrict__ z2, const float* __restrict__ wmu,
                        const float* __restrict__ bmu, const float* __restrict__ wls,
                        const float* __restrict__ bls, float* __restrict__ out) {
  int i = blockIdx.x * blockDim.x + threadIdx.x;
  if (i >= BROWS * OUTF) return;
  int b = i / OUTF, o = i % OUTF;
  const float* zr = z2 + (size_t)b * OUTF;
  float mu = bmu[o], ls = bls[o];
  for (int k = 0; k < OUTF; ++k) {
    float zv = zr[k];
    mu += zv * wmu[k * OUTF + o];
    ls += zv * wls[k * OUTF + o];
  }
  out[i] = mu;
  out[(size_t)BROWS * OUTF + i] = ls;
}

// pass edge_index through as third output (raw int bits)
__global__ void k_copy_edges(const int* __restrict__ ei, int* __restrict__ out_i) {
  int i = blockIdx.x * blockDim.x + threadIdx.x;
  if (i >= 2 * EIN) return;
  out_i[2 * BROWS * OUTF + i] = ei[i];
}

// ---------------- host launch ----------------
extern "C" void kernel_launch(void* const* d_in, const int* in_sizes, int n_in,
                              void* d_out, int out_size, void* d_ws, size_t ws_size,
                              hipStream_t stream) {
  const float* x     = (const float*)d_in[0];
  const int*   ei    = (const int*)  d_in[1];
  const float* wgat  = (const float*)d_in[2];
  const float* asrc  = (const float*)d_in[3];
  const float* adst  = (const float*)d_in[4];
  const float* bgat  = (const float*)d_in[5];
  const float* wht   = (const float*)d_in[6];
  const float* bht   = (const float*)d_in[7];
  const float* g1    = (const float*)d_in[8];
  const float* be1   = (const float*)d_in[9];
  const float* wgcn  = (const float*)d_in[10];
  // d_in[11] = b_gcn: mathematically cancels inside BatchNorm (mean subtraction)
  const float* g2    = (const float*)d_in[12];
  const float* be2   = (const float*)d_in[13];
  const float* wl1   = (const float*)d_in[14];
  const float* bl1   = (const float*)d_in[15];
  const float* wl2   = (const float*)d_in[16];
  const float* bl2   = (const float*)d_in[17];
  const float* wtr   = (const float*)d_in[18];
  const float* btr   = (const float*)d_in[19];
  const float* wmu   = (const float*)d_in[20];
  const float* bmu   = (const float*)d_in[21];
  const float* wls   = (const float*)d_in[22];
  const float* bls   = (const float*)d_in[23];

  float*    ws    = (float*)d_ws;
  float*    h     = ws + OFF_H;
  float*    a_s   = ws + OFF_AS;
  float*    a_d   = ws + OFF_AD;
  unsigned* mkey  = (unsigned*)(ws + OFF_MK);
  float*    ssum  = ws + OFF_SS;
  float*    deg   = ws + OFF_DEG;
  float*    agg   = ws + OFF_AGG;
  float*    xg    = ws + OFF_XG;
  float*    hc    = ws + OFF_HC;
  float*    xc    = ws + OFF_XC;
  float*    dinv  = ws + OFF_DIN;
  float*    z     = ws + OFF_Z;
  float*    z2    = ws + OFF_Z2;
  float*    mean1 = ws + OFF_BN;
  float*    is1   = mean1 + E1F;
  float*    mean2 = is1 + E1F;
  float*    is2   = mean2 + E2F;
  float*    out   = (float*)d_out;

  const int TB = 256;
  const int gN   = NN / TB;                  // node-parallel grid
  const int gN3  = (NN * HEADS) / TB;
  const int gE   = (ETOT + TB - 1) / TB;     // edge-parallel grid
  const int gGAT = NN / (16 * 8);            // 1 wave / 16 rows, 8 waves / block
  const int gTR  = ((BROWS / 16) * (OUTF / 16)) / 8;

  // zero accumulators: [mkey|ssum|deg|agg] are contiguous; xc separate
  hipLaunchKernelGGL(k_zero, dim3(4096), dim3(TB), 0, stream,
                     ws + OFF_MK, (size_t)NN * (HEADS * 2) + NN + (size_t)NN * HF);
  hipLaunchKernelGGL(k_zero, dim3(2048), dim3(TB), 0, stream, xc, (size_t)NN * E2F);

  // GAT
  hipLaunchKernelGGL(k_gat_proj, dim3(gGAT), dim3(TB), 0, stream, x, wgat, h);
  hipLaunchKernelGGL(k_attn,     dim3(gN3),  dim3(TB), 0, stream, h, asrc, adst, a_s, a_d);
  hipLaunchKernelGGL(k_edge_max, dim3(gE),   dim3(TB), 0, stream, ei, a_s, a_d, mkey, deg);
  hipLaunchKernelGGL(k_edge_sum, dim3(gE),   dim3(TB), 0, stream, ei, a_s, a_d, mkey, ssum);
  hipLaunchKernelGGL(k_edge_agg, dim3(gE),   dim3(TB), 0, stream, ei, a_s, a_d, mkey, ssum, h, agg);

  // head_transform + BN1
  hipLaunchKernelGGL(k_node_ht,   dim3(gN),  dim3(TB), 0, stream, agg, bgat, wht, bht, xg);
  hipLaunchKernelGGL(k_bn_reduce, dim3(E1F), dim3(TB), 0, stream, xg, E1F, mean1, is1);

  // GCN
  hipLaunchKernelGGL(k_node_gcn, dim3(gN), dim3(TB), 0, stream, xg, mean1, is1, g1, be1, wgcn, hc);
  hipLaunchKernelGGL(k_dinv,     dim3(gN), dim3(TB), 0, stream, deg, dinv);
  hipLaunchKernelGGL(k_edge_gcn, dim3(gE), dim3(TB), 0, stream, ei, dinv, hc, xc);
  hipLaunchKernelGGL(k_bn_reduce, dim3(E2F), dim3(TB), 0, stream, xc, E2F, mean2, is2);

  // fused per-node MLP 9 -> 100 -> 4
  hipLaunchKernelGGL(k_node_mlp, dim3(gN), dim3(TB), 0, stream,
                     xc, mean2, is2, g2, be2, wl1, bl1, wl2, bl2, z);

  // readout GEMM (WMMA) + heads
  hipLaunchKernelGGL(k_trans_gemm, dim3(gTR), dim3(TB), 0, stream, z, wtr, btr, z2);
  hipLaunchKernelGGL(k_heads, dim3((BROWS * OUTF) / TB), dim3(TB), 0, stream,
                     z2, wmu, bmu, wls, bls, out);

  // edge_index passthrough
  hipLaunchKernelGGL(k_copy_edges, dim3((2 * EIN) / TB), dim3(TB), 0, stream,
                     ei, (int*)d_out);
}
